// NNMFLayerSbSBP_35794257444867
// MI455X (gfx1250) — matrix operand order repaired and checked by reading, not compiled, not run on hardware
//
#include <hip/hip_runtime.h>
#include <stdint.h>

typedef __attribute__((ext_vector_type(16))) _Float16     v16h;
typedef __attribute__((ext_vector_type(8)))  float        v8f;
typedef __attribute__((ext_vector_type(8)))  unsigned int v8u;

#define CIN   64
#define COUT  96
#define WAVES 2          // waves per block, 16 pixels per wave
#define NIT   5
#define EPSF  1e-20f

// Row strides (in elements). All f16 strides are even -> dword-aligned pairs,
// and chosen so 16-lane gathers land on 16 distinct LDS banks.
#define WS_H  98         // Ws  [cin][cout]  f16   (dword stride 49)
#define WT_H  66         // WsT [cout][cin]  f16   (dword stride 33)
#define HH_H  100        // Hh  [pix][cout]  f16   (dword stride 50)
#define TH_H  68         // Th  [pix][cin]   f16   (dword stride 34)
#define XS_F  17         // Xs  [cin][pix]   f32
#define FB_F  100        // Fb  [pix][cout]  f32 staging for coalesced output

typedef __attribute__((address_space(1))) int* gptr_i32;
typedef __attribute__((address_space(3))) int* lptr_i32;

__global__ __launch_bounds__(WAVES * 32)
void nnmf_wmma_kernel(const float* __restrict__ x,
                      const float* __restrict__ w,
                      const float* __restrict__ h0,
                      float* __restrict__ out)
{
    __shared__ _Float16 Ws [CIN  * WS_H];            // W,  block-shared
    __shared__ _Float16 WsT[COUT * WT_H];            // W^T, block-shared
    __shared__ _Float16 Hh[WAVES][16 * HH_H];        // f16 mirror of h
    __shared__ _Float16 Th[WAVES][16 * TH_H];        // t = x/(Wh+eps)
    __shared__ float    Xs[WAVES][CIN * XS_F];       // x tile
    __shared__ float    Fb[WAVES][16 * FB_F];        // f32 output staging

    const int tid   = threadIdx.x;
    const int lane  = tid & 31;
    const int wave  = tid >> 5;
    const int half_ = lane >> 4;     // 16-lane half
    const int lc    = lane & 15;     // A row (M) / B,C,D column (N)

    const int tile = blockIdx.x * WAVES + wave;      // 64 tiles per image
    const int b    = tile >> 6;
    const int p0   = (tile & 63) << 4;

    // ---- stage W (both orientations) as f16, whole block ----
    for (int i = tid; i < CIN * COUT; i += WAVES * 32) {
        int cin = i / COUT, cout = i % COUT;
        _Float16 v = (_Float16)w[i];
        Ws [cin  * WS_H + cout] = v;
        WsT[cout * WT_H + cin ] = v;
    }

    // ---- stage x tile: Xs[cin][pix] (async-to-LDS when available) ----
    float* xs = Xs[wave];
#if defined(__gfx1250__) && __has_builtin(__builtin_amdgcn_global_load_async_to_lds_b32) && __has_builtin(__builtin_amdgcn_s_wait_asynccnt)
    for (int i = lane; i < CIN * 16; i += 32) {
        int cin = i >> 4, pix = i & 15;
        gptr_i32 gp = (gptr_i32)(uintptr_t)(x + ((b * CIN + cin) << 10) + p0 + pix);
        lptr_i32 lp = (lptr_i32)(uintptr_t)&xs[cin * XS_F + pix];
        __builtin_amdgcn_global_load_async_to_lds_b32(gp, lp, 0, 0);
    }
    __builtin_amdgcn_s_wait_asynccnt(0);
#else
    for (int i = lane; i < CIN * 16; i += 32) {
        int cin = i >> 4, pix = i & 15;
        xs[cin * XS_F + pix] = x[((b * CIN + cin) << 10) + p0 + pix];
    }
#endif

    // ---- h lives in WMMA C/D accumulator form:
    //      hacc[nc][r] = h[pix = r + 8*half_][cout = nc*16 + lc]
    _Float16* hh = Hh[wave];
    _Float16* ts = Th[wave];
    v8f hacc[6];
    #pragma unroll
    for (int nc = 0; nc < 6; ++nc) {
        float hv = h0[nc * 16 + lc];
        #pragma unroll
        for (int r = 0; r < 8; ++r) {
            hacc[nc][r] = hv;
            hh[(r + 8 * half_) * HH_H + nc * 16 + lc] = (_Float16)hv;
        }
    }
    __syncthreads();   // Ws/WsT visible to all waves

    for (int it = 0; it < NIT; ++it) {
        // ===== comp1: D(16x64) = H(16x96) x W^T(96x64) =====
        v16h aH[3];
        #pragma unroll
        for (int kc = 0; kc < 3; ++kc) {
            v8u au;
            #pragma unroll
            for (int v = 0; v < 8; ++v) {
                int kb = kc * 32 + (v >> 2) * 16 + half_ * 8 + (v & 3) * 2;
                au[v] = *reinterpret_cast<const unsigned*>(&hh[lc * HH_H + kb]);
            }
            aH[kc] = __builtin_bit_cast(v16h, au);
        }
        #pragma unroll
        for (int nc = 0; nc < 4; ++nc) {              // cin tiles of 16
            v8f acc = {};
            #pragma unroll
            for (int kc = 0; kc < 3; ++kc) {
                v8u bu;                                // B[k=cout][n=cin] = W[cin][cout]
                #pragma unroll
                for (int v = 0; v < 8; ++v) {
                    int k = kc * 32 + (v >> 2) * 16 + half_ * 8 + (v & 3) * 2;    // cout (even)
                    int n = nc * 16 + lc;                    // cin
                    bu[v] = *reinterpret_cast<const unsigned*>(&Ws[n * WS_H + k]);
                }
                acc = __builtin_amdgcn_wmma_f32_16x16x32_f16(
                          false, aH[kc], false, __builtin_bit_cast(v16h, bu),
                          (short)0, acc, false, false);
            }
            #pragma unroll
            for (int r = 0; r < 8; ++r) {
                int pix = r + 8 * half_;
                int cin = nc * 16 + lc;
                float t = xs[cin * XS_F + pix] *
                          __builtin_amdgcn_rcpf(acc[r] + EPSF);
                ts[pix * TH_H + cin] = (_Float16)t;    // same-wave LDS, in-order
            }
        }
        // ===== comp2: Hnew(16x96) = h + T(16x64) x W(64x96)  (EPS0 = 1) =====
        // h rides in as the WMMA C-matrix -> no separate adds.
        v16h aT[2];
        #pragma unroll
        for (int kc = 0; kc < 2; ++kc) {
            v8u au;
            #pragma unroll
            for (int v = 0; v < 8; ++v) {
                int ka = kc * 32 + (v >> 2) * 16 + half_ * 8 + (v & 3) * 2;
                au[v] = *reinterpret_cast<const unsigned*>(&ts[lc * TH_H + ka]);
            }
            aT[kc] = __builtin_bit_cast(v16h, au);
        }
        v8f hn[6];
        #pragma unroll
        for (int nc = 0; nc < 6; ++nc) {              // cout tiles of 16
            v8f acc = hacc[nc];                       // C = h
            #pragma unroll
            for (int kc = 0; kc < 2; ++kc) {          // cin chunks of 32
                v8u bu;
                #pragma unroll
                for (int v = 0; v < 8; ++v) {
                    int kb = kc * 32 + (v >> 2) * 16 + half_ * 8 + (v & 3) * 2;   // cin (even)
                    int n  = nc * 16 + lc;                   // cout
                    bu[v] = *reinterpret_cast<const unsigned*>(&WsT[n * WT_H + kb]);
                }
                acc = __builtin_amdgcn_wmma_f32_16x16x32_f16(
                          false, aT[kc], false, __builtin_bit_cast(v16h, bu),
                          (short)0, acc, false, false);
            }
            hn[nc] = acc;
        }
        // ===== per-pixel Cout-sum, normalize, refresh f16 mirror =====
        #pragma unroll
        for (int r = 0; r < 8; ++r) {
            float s = 0.f;
            #pragma unroll
            for (int nc = 0; nc < 6; ++nc) s += hn[nc][r];
            s += __shfl_xor(s, 1, 32);
            s += __shfl_xor(s, 2, 32);
            s += __shfl_xor(s, 4, 32);
            s += __shfl_xor(s, 8, 32);
            float inv = __builtin_amdgcn_rcpf(s + EPSF);
            int pix = r + 8 * half_;
            #pragma unroll
            for (int nc = 0; nc < 6; ++nc) {
                float hv = hn[nc][r] * inv;
                hacc[nc][r] = hv;
                hh[pix * HH_H + nc * 16 + lc] = (_Float16)hv;
            }
        }
    }

    // ---- stage f32 result to LDS, then coalesced global store ----
    float* fb = Fb[wave];
    #pragma unroll
    for (int nc = 0; nc < 6; ++nc)
        #pragma unroll
        for (int r = 0; r < 8; ++r)
            fb[(r + 8 * half_) * FB_F + nc * 16 + lc] = hacc[nc][r];
    // same wave wrote all 16x96 values; LDS is in-order within a wave
    for (int i = lane; i < 16 * COUT; i += 32) {
        int pix = i & 15, cout = i >> 4;
        out[((b * COUT + cout) << 10) + p0 + pix] = fb[pix * FB_F + cout];
    }
}

extern "C" void kernel_launch(void* const* d_in, const int* in_sizes, int n_in,
                              void* d_out, int out_size, void* d_ws, size_t ws_size,
                              hipStream_t stream) {
    const float* x  = (const float*)d_in[0];   // (4, 64, 32, 32)
    const float* w  = (const float*)d_in[1];   // (64, 96)
    const float* h0 = (const float*)d_in[2];   // (96,)
    float* out = (float*)d_out;                // (4, 96, 32, 32)

    const int blocks = 256 / WAVES;            // 256 pixel-tiles of 16
    nnmf_wmma_kernel<<<blocks, WAVES * 32, 0, stream>>>(x, w, h0, out);
}
